// Policy_79920751444239
// MI455X (gfx1250) — compile-verified, matrix operand order checked by reading, not run on hardware
//
#include <hip/hip_runtime.h>

typedef __attribute__((ext_vector_type(16))) _Float16 h16;
typedef __attribute__((ext_vector_type(8)))  _Float16 h8;
typedef __attribute__((ext_vector_type(4)))  _Float16 h4;
typedef __attribute__((ext_vector_type(8)))  float    f8;

#define BS_   16384
#define A_    32
#define F_    256
#define D_    64
#define EPSL  1e-6f
#define XS    264   // x LDS row stride in halfs (256 + 8 pad -> conflict-free A loads)
#define KVS   68    // k/v LDS row stride in floats (64 + 4 pad)

// ---------------------------------------------------------------------------
// Prep kernel 1: pack Wq/Wk/Wv/Wfc (each 64x256 f32, row-major over K) into
// f16 WMMA B-fragments.  Fragment layout for V_WMMA_F32_16X16X32_F16 B (KxN =
// 32x16), mirroring the ISA's 16-bit A layout with M->N:
//   lanes 0-15 : N = lane,     halfs 0-7 -> K = 0..7,  halfs 8-15 -> K = 16..23
//   lanes 16-31: N = lane-16,  halfs 0-7 -> K = 8..15, halfs 8-15 -> K = 24..31
// Storage: frags[w][nt][ks][lane][16 halfs], w in {Wq,Wk,Wv,Wfc}.
// ---------------------------------------------------------------------------
__global__ void prep_frags(const float* __restrict__ Wq, const float* __restrict__ Wk,
                           const float* __restrict__ Wv, const float* __restrict__ Wfc,
                           _Float16* __restrict__ frags) {
  int idx = blockIdx.x * 256 + threadIdx.x;          // 4 * 16384 total
  int h    = idx & 15;
  int lane = (idx >> 4) & 31;
  int fi   = (idx >> 9) & 31;                        // nt*8 + ks
  int ks   = fi & 7;
  int nt   = fi >> 3;
  int w    = idx >> 14;
  const float* W = (w == 0) ? Wq : (w == 1) ? Wk : (w == 2) ? Wv : Wfc;
  int l15 = lane & 15;
  int hi  = lane >> 4;
  int N   = nt * 16 + l15;
  int kis = (h < 8) ? (hi * 8 + h) : (16 + hi * 8 + (h - 8));
  int K   = ks * 32 + kis;
  frags[idx] = (_Float16)W[N * F_ + K];
}

// Prep kernel 2: transpose W1, W2 (64x64) so the MLP inner loop reads
// coalesced columns.
__global__ void prep_transpose(const float* __restrict__ W1, const float* __restrict__ W2,
                               float* __restrict__ W1T, float* __restrict__ W2T) {
  int idx = blockIdx.x * 256 + threadIdx.x;          // 2 * 4096 total
  int w = idx >> 12;
  int e = idx & 4095;
  int i = e >> 6;
  int d = e & 63;
  if (w == 0) W1T[d * 64 + i] = W1[i * 64 + d];
  else        W2T[d * 64 + i] = W2[i * 64 + d];
}

// ---------------------------------------------------------------------------
// Main fused kernel: one 128-thread block (4 wave32) per batch row.
// ---------------------------------------------------------------------------
__global__ __launch_bounds__(128) void fused_policy(
    const int*   __restrict__ agent,
    const float* __restrict__ x,
    const _Float16* __restrict__ frags,
    const float* __restrict__ bq, const float* __restrict__ bk,
    const float* __restrict__ bv, const float* __restrict__ bfc,
    const float* __restrict__ gq, const float* __restrict__ betq,
    const float* __restrict__ gk, const float* __restrict__ betk,
    const float* __restrict__ gv, const float* __restrict__ betv,
    const float* __restrict__ b1, const float* __restrict__ b2,
    const float* __restrict__ W1T, const float* __restrict__ W2T,
    float* __restrict__ out)
{
  __shared__ __align__(16) _Float16 xlds[48 * XS];   // rows 0-30 ally, 31 zero, 32 self, 33-47 zero
  __shared__ float klds[32 * KVS];
  __shared__ float vlds[32 * KVS];
  __shared__ float qlds[64];
  __shared__ float attnw[32];
  __shared__ float olds[64];
  __shared__ float hlds[64];

  const int tid = threadIdx.x;
  const int b   = blockIdx.x;
  const int aid = agent[0];

  // ---- Phase 0: stage input row, f32 -> f16, reorder agents ----
  const float4* xrow = (const float4*)(x + (size_t)b * (A_ * F_));
  for (int i = tid; i < (A_ * F_) / 4; i += 128) {
    float4 v = xrow[i];
    int fidx = i * 4;
    int a    = fidx >> 8;
    int col  = fidx & 255;
    int row  = (a == aid) ? 32 : (a < aid ? a : a - 1);
    h4 hv = { (_Float16)v.x, (_Float16)v.y, (_Float16)v.z, (_Float16)v.w };
    *(h4*)&xlds[row * XS + col] = hv;
  }
  for (int i = tid; i < 16 * XS; i += 128) {      // zero pad rows 31, 33..47
    int ri = i / XS, c = i % XS;
    int row = (ri == 0) ? 31 : (32 + ri);
    xlds[row * XS + c] = (_Float16)0.f;
  }
  __syncthreads();

  const int wave = tid >> 5;
  const int lane = tid & 31;
  const int l15  = lane & 15;
  const int halfid = lane >> 4;

  // ---- Phase 1: WMMA projections (6 tile-tasks over 4 waves) ----
  // t0/t1: K rows 0-15 / 16-31; t2/t3: V; t4: Q (row 0); t5: Wfc (row 0 -> out)
  for (int t = wave; t < 6; t += 4) {
    int arow; const _Float16* fw; const float* bias; const float* g; const float* be;
    switch (t) {
      case 0:  arow = 0;  fw = frags + 16384; bias = bk;  g = gk; be = betk; break;
      case 1:  arow = 16; fw = frags + 16384; bias = bk;  g = gk; be = betk; break;
      case 2:  arow = 0;  fw = frags + 32768; bias = bv;  g = gv; be = betv; break;
      case 3:  arow = 16; fw = frags + 32768; bias = bv;  g = gv; be = betv; break;
      case 4:  arow = 32; fw = frags;         bias = bq;  g = gq; be = betq; break;
      default: arow = 32; fw = frags + 49152; bias = bfc; g = nullptr; be = nullptr; break;
    }
    f8 acc[4];
    #pragma unroll
    for (int nt = 0; nt < 4; ++nt) {
      float bval = bias[nt * 16 + l15];
      #pragma unroll
      for (int r = 0; r < 8; ++r) acc[nt][r] = bval;
    }
    #pragma unroll
    for (int ks = 0; ks < 8; ++ks) {
      // A fragment: lanes 0-15 row M=l15 K=32ks+{0..7,16..23}; lanes 16-31 +8
      const _Float16* ap = &xlds[(arow + l15) * XS + ks * 32 + halfid * 8];
      h8 alo = *(const h8*)ap;
      h8 ahi = *(const h8*)(ap + 16);
      h16 afrag = __builtin_shufflevector(alo, ahi, 0,1,2,3,4,5,6,7,8,9,10,11,12,13,14,15);
      #pragma unroll
      for (int nt = 0; nt < 4; ++nt) {
        h16 bfrag = *(const h16*)(fw + (size_t)((nt * 8 + ks) * 32 + lane) * 16);
        acc[nt] = __builtin_amdgcn_wmma_f32_16x16x32_f16(
            false, afrag, false, bfrag, (short)0, acc[nt], false, false);
      }
    }
    if (t < 5) {
      // LayerNorm over N=64 per row.  C layout: row M = halfid*8 + r,
      // N = nt*16 + l15; reduce across the 16 lanes of each half-wave.
      float g4[4], e4[4];
      #pragma unroll
      for (int nt = 0; nt < 4; ++nt) { g4[nt] = g[nt * 16 + l15]; e4[nt] = be[nt * 16 + l15]; }
      #pragma unroll
      for (int r = 0; r < 8; ++r) {
        float s = acc[0][r] + acc[1][r] + acc[2][r] + acc[3][r];
        s += __shfl_xor(s, 1, 16); s += __shfl_xor(s, 2, 16);
        s += __shfl_xor(s, 4, 16); s += __shfl_xor(s, 8, 16);
        float mean = s * (1.f / 64.f);
        float vs = 0.f;
        #pragma unroll
        for (int nt = 0; nt < 4; ++nt) { float d = acc[nt][r] - mean; vs += d * d; }
        vs += __shfl_xor(vs, 1, 16); vs += __shfl_xor(vs, 2, 16);
        vs += __shfl_xor(vs, 4, 16); vs += __shfl_xor(vs, 8, 16);
        float inv = rsqrtf(vs * (1.f / 64.f) + EPSL);
        #pragma unroll
        for (int nt = 0; nt < 4; ++nt)
          acc[nt][r] = (acc[nt][r] - mean) * inv * g4[nt] + e4[nt];
      }
    }
    if (t < 4) {
      float* dst = (t < 2 ? klds : vlds) + (t & 1) * 16 * KVS;
      #pragma unroll
      for (int r = 0; r < 8; ++r)
        #pragma unroll
        for (int nt = 0; nt < 4; ++nt)
          dst[(halfid * 8 + r) * KVS + nt * 16 + l15] = acc[nt][r];
    } else if (t == 4) {
      if (lane < 16)
        #pragma unroll
        for (int nt = 0; nt < 4; ++nt) qlds[nt * 16 + lane] = acc[nt][0];
    } else {
      if (lane < 16)
        #pragma unroll
        for (int nt = 0; nt < 4; ++nt)
          out[(size_t)b * 128 + 64 + nt * 16 + lane] = acc[nt][0];
    }
  }
  __syncthreads();

  // ---- Phase 2: scores + softmax over 31 allies (wave 0) ----
  if (wave == 0) {
    float s = 0.f;
    if (lane < 31) {
      #pragma unroll 4
      for (int d = 0; d < 64; ++d) s += qlds[d] * klds[lane * KVS + d];
      s *= 0.125f;                                   // 1 / sqrt(D)
    }
    float sm = (lane < 31) ? s : -3.4e38f;
    for (int m = 16; m >= 1; m >>= 1) sm = fmaxf(sm, __shfl_xor(sm, m, 32));
    float e = (lane < 31) ? __expf(s - sm) : 0.f;
    float tot = e;
    for (int m = 16; m >= 1; m >>= 1) tot += __shfl_xor(tot, m, 32);
    attnw[lane] = e / tot;
  }
  __syncthreads();

  // ---- Phase 3: o = attn @ v ----
  if (tid < 64) {
    float o = 0.f;
    #pragma unroll 4
    for (int j = 0; j < 31; ++j) o += attnw[j] * vlds[j * KVS + tid];
    olds[tid] = o;
  }
  __syncthreads();

  // ---- Phase 4: MLP 64->64->64 ----
  if (tid < 64) {
    float h = b1[tid];
    #pragma unroll 4
    for (int d = 0; d < 64; ++d) h += olds[d] * W1T[d * 64 + tid];
    hlds[tid] = fmaxf(h, 0.f);
  }
  __syncthreads();
  if (tid < 64) {
    float y = b2[tid];
    #pragma unroll 4
    for (int d = 0; d < 64; ++d) y += hlds[d] * W2T[d * 64 + tid];
    out[(size_t)b * 128 + tid] = y;
  }
}

extern "C" void kernel_launch(void* const* d_in, const int* in_sizes, int n_in,
                              void* d_out, int out_size, void* d_ws, size_t ws_size,
                              hipStream_t stream) {
  const int*   agent = (const int*)  d_in[0];
  const float* x     = (const float*)d_in[1];
  const float* Wq    = (const float*)d_in[2];
  const float* bq    = (const float*)d_in[3];
  const float* Wk    = (const float*)d_in[4];
  const float* bk    = (const float*)d_in[5];
  const float* Wv    = (const float*)d_in[6];
  const float* bv    = (const float*)d_in[7];
  const float* gq    = (const float*)d_in[8];
  const float* betq  = (const float*)d_in[9];
  const float* gk    = (const float*)d_in[10];
  const float* betk  = (const float*)d_in[11];
  const float* gv    = (const float*)d_in[12];
  const float* betv  = (const float*)d_in[13];
  const float* W1    = (const float*)d_in[14];
  const float* b1    = (const float*)d_in[15];
  const float* W2    = (const float*)d_in[16];
  const float* b2    = (const float*)d_in[17];
  const float* Wfc   = (const float*)d_in[18];
  const float* bfc   = (const float*)d_in[19];

  // Workspace layout: [0,128KB) f16 B-fragments for Wq/Wk/Wv/Wfc,
  // [128KB,144KB) W1T, [144KB,160KB) W2T.
  _Float16* frags = (_Float16*)d_ws;
  float*    W1T   = (float*)((char*)d_ws + 131072);
  float*    W2T   = (float*)((char*)d_ws + 147456);

  prep_frags<<<256, 256, 0, stream>>>(Wq, Wk, Wv, Wfc, frags);
  prep_transpose<<<32, 256, 0, stream>>>(W1, W2, W1T, W2T);

  int rows = in_sizes[1] / (A_ * F_);   // 16384
  fused_policy<<<rows, 128, 0, stream>>>(
      agent, x, frags, bq, bk, bv, bfc,
      gq, betq, gk, betk, gv, betv,
      b1, b2, W1T, W2T, (float*)d_out);
}
